// SdfVAE_48653389529166
// MI455X (gfx1250) — matrix operand order repaired and checked by reading, not compiled, not run on hardware
//
#include <hip/hip_runtime.h>
#include <cstdint>

// ---------------------------------------------------------------------------
// Types / helpers
// ---------------------------------------------------------------------------
typedef __attribute__((ext_vector_type(16))) __bf16 v16bf;
typedef __attribute__((ext_vector_type(8)))  __bf16 v8bf;
typedef __attribute__((ext_vector_type(8)))  float  v8f;

#define DEVINL __device__ __forceinline__

DEVINL __bf16 bf_bits(unsigned short u) { __bf16 h; __builtin_memcpy(&h, &u, 2); return h; }

DEVINL __bf16 f2bf(float f) {
    union { float f; unsigned u; } v; v.f = f;
    unsigned r = v.u + 0x7FFFu + ((v.u >> 16) & 1u);   // round-to-nearest-even
    return bf_bits((unsigned short)(r >> 16));
}
DEVINL float bf2f(__bf16 h) {
    unsigned short s; __builtin_memcpy(&s, &h, 2);
    union { unsigned u; float f; } v; v.u = ((unsigned)s) << 16;
    return v.f;
}
DEVINL float gelu_f(float x) { return 0.5f * x * (1.f + erff(x * 0.70710678118654752f)); }

DEVINL v8f wmma_bf16(v16bf a, v16bf b, v8f c) {
    // D = A(16x32 bf16) x B(32x16 bf16) + C(16x16 f32)
    return __builtin_amdgcn_wmma_f32_16x16x32_bf16(false, a, false, b, (short)0, c, false, false);
}

// ---------------------------------------------------------------------------
// Constants
// ---------------------------------------------------------------------------
#define Bsz   4
#define Npts  16384
#define R3    32768
#define Cch   128
#define NPTOT (Bsz * Npts)          // 65536

// ---------------------------------------------------------------------------
// Weight repack kernels: build exact per-lane WMMA B-fragments.
// B 32x16 layout: element h of lane half g holds K = 8g + h (h<8), 16 + 8g + (h-8) (h>=8),
// N = lane&15.  Packed index = ((kt*Ntiles + nt)*32 + lane)*16 + h.
// ---------------------------------------------------------------------------
__global__ __launch_bounds__(256) void repack_conv_w(const float* __restrict__ w,
                                                     __bf16* __restrict__ pk,
                                                     int CIN, int COUT) {
    size_t i = (size_t)blockIdx.x * 256 + threadIdx.x;
    size_t total = (size_t)27 * CIN * COUT;
    if (i >= total) return;
    int h    = (int)(i & 15);
    int lane = (int)((i >> 4) & 31);
    size_t tile = i >> 9;
    int Nt = COUT >> 4;
    int nt = (int)(tile % Nt);
    int kt = (int)(tile / Nt);
    int k = kt * 32 + ((lane >> 4) << 3) + (h & 7) + ((h >> 3) << 4);  // K within [kt*32, kt*32+32)
    int n = nt * 16 + (lane & 15);
    int tap = k / CIN, ci = k % CIN;           // k-major ordering: k = tap*CIN + ci
    // w is OIDHW: (COUT, CIN, 3,3,3) -> w[n][ci][tap]
    pk[i] = f2bf(w[((size_t)n * CIN + ci) * 27 + tap]);
}

__global__ __launch_bounds__(256) void repack_mlp_w(const float* __restrict__ w,
                                                    __bf16* __restrict__ pk,
                                                    int K, int N) {
    size_t i = (size_t)blockIdx.x * 256 + threadIdx.x;
    size_t total = (size_t)K * N;
    if (i >= total) return;
    int h    = (int)(i & 15);
    int lane = (int)((i >> 4) & 31);
    size_t tile = i >> 9;
    int Nt = N >> 4;
    int nt = (int)(tile % Nt);
    int kt = (int)(tile / Nt);
    int k = kt * 32 + ((lane >> 4) << 3) + (h & 7) + ((h >> 3) << 4);
    int n = nt * 16 + (lane & 15);
    pk[i] = f2bf(w[(size_t)k * N + n]);        // w is (K, N) row-major
}

// ---------------------------------------------------------------------------
// Point MLP (4 -> 64 gelu -> 128) + scatter-add into fp32 grid (ch-last)
// ---------------------------------------------------------------------------
__global__ __launch_bounds__(256) void point_mlp_scatter(
    const float* __restrict__ pts, const float* __restrict__ sdf,
    const float* __restrict__ w0, const float* __restrict__ b0,
    const float* __restrict__ w1, const float* __restrict__ b1,
    float* __restrict__ grid) {
    __shared__ float sw1[64 * 128];
    __shared__ float sw0[4 * 64];
    __shared__ float sb0[64];
    __shared__ float sb1[128];
    for (int i = threadIdx.x; i < 64 * 128; i += 256) sw1[i] = w1[i];
    sw0[threadIdx.x] = w0[threadIdx.x];
    if (threadIdx.x < 64)  sb0[threadIdx.x] = b0[threadIdx.x];
    if (threadIdx.x < 128) sb1[threadIdx.x] = b1[threadIdx.x];
    __syncthreads();

    int p = blockIdx.x * 256 + threadIdx.x;    // 0 .. 65535
    int b = p >> 14;
    float x0 = pts[p * 3 + 0], x1 = pts[p * 3 + 1], x2 = pts[p * 3 + 2], x3 = sdf[p];

    float h[64];
#pragma unroll
    for (int j = 0; j < 64; ++j) {
        float s = sb0[j] + x0 * sw0[j] + x1 * sw0[64 + j] + x2 * sw0[128 + j] + x3 * sw0[192 + j];
        h[j] = gelu_f(s);
    }
    int c0 = (int)fminf(fmaxf((x0 + 1.f) * 0.5f * 31.f, 0.f), 31.f);
    int c1 = (int)fminf(fmaxf((x1 + 1.f) * 0.5f * 31.f, 0.f), 31.f);
    int c2 = (int)fminf(fmaxf((x2 + 1.f) * 0.5f * 31.f, 0.f), 31.f);
    size_t base = ((size_t)b * R3 + c0 * 1024 + c1 * 32 + c2) * Cch;
    for (int c = 0; c < Cch; ++c) {
        float s = sb1[c];
#pragma unroll
        for (int j = 0; j < 64; ++j) s += h[j] * sw1[j * 128 + c];
        atomicAdd(&grid[base + c], s);
    }
}

__global__ __launch_bounds__(256) void cvt_f32_bf16(const float* __restrict__ src,
                                                    __bf16* __restrict__ dst, size_t n) {
    size_t i = (size_t)blockIdx.x * 256 + threadIdx.x;
    if (i < n) dst[i] = f2bf(src[i]);
}

// ---------------------------------------------------------------------------
// Implicit-GEMM 3x3x3 conv, SAME padding, channels-last bf16 activations.
// Software-pipelined with a 2-stage ping-pong register buffer (loop unrolled
// by 2), so there are no inter-stage register copies: loads for step s+1 are
// in flight while the WMMAs of step s execute; s_wait_loadcnt only waits one
// stage back.  Halo handling is branch-free: invalid lanes read from a zeroed
// page (address select), keeping EXEC all-ones for WMMA.
// MODE 0: bf16 ch-last out.  MODE 1: fp32 NCDHW out (mean/logvar -> d_out).
// ---------------------------------------------------------------------------
template <int CIN, int COUT, int MODE>
__global__ __launch_bounds__(256) void conv3d_wmma(
    const __bf16* __restrict__ in, const __bf16* __restrict__ W,
    const float* __restrict__ bias, const __bf16* __restrict__ zpage,
    __bf16* __restrict__ outb, float* __restrict__ outf) {
    constexpr int WN = COUT / 64;       // n-groups per block
    constexpr int CSTEPS = CIN / 32;    // k-steps per tap (4 or 8, power of 2)
    constexpr int KT = 27 * CSTEPS;     // total k-steps (108 / 216: even)
    int wave = threadIdx.x >> 5, lane = threadIdx.x & 31;
    int mtile = blockIdx.x * (8 / WN) + wave / WN;
    int ngrp  = wave % WN;
    int m0 = mtile * 16;
    int x0 = m0 & 31;
    int y  = (m0 >> 5) & 31;
    int z  = (m0 >> 10) & 31;
    int b  = m0 >> 15;
    int xm = x0 + (lane & 15);
    int off = (lane >> 4) * 8;

    const __bf16* base  = in + (size_t)b * (32 * 32 * 32) * CIN;
    const __bf16* wbase = W + (size_t)(ngrp * 4) * 512 + lane * 16;

    // per-step A address: step s -> (tap, c0); invalid lanes -> zero page
    auto aptr = [&](int s) -> const __bf16* {
        int tap = s / CSTEPS;                       // power-of-2 divide
        int c0  = (s % CSTEPS) * 32;
        int dz = tap / 9;
        int r9 = tap - dz * 9;
        int dy = r9 / 3;
        int dx = r9 - dy * 3;
        int zz = z + dz - 1, yy = y + dy - 1, xx = xm + dx - 1;
        bool valid = ((unsigned)zz < 32u) & ((unsigned)yy < 32u) & ((unsigned)xx < 32u);
        const __bf16* p =
            base + ((((zz & 31) * 32 + (yy & 31)) * 32) + (xx & 31)) * CIN + c0 + off;
        return valid ? p : zpage;
    };

    v8f acc[4] = {};
    // 2-stage ping-pong buffers (constant-indexed -> stay in registers)
    v8bf alo[2], ahi[2];
    v16bf bb[2][4];
    {   // prologue: stage 0 <- step 0
        const __bf16* p = aptr(0);
        alo[0] = *(const v8bf*)p;
        ahi[0] = *(const v8bf*)(p + 16);
#pragma unroll
        for (int j = 0; j < 4; ++j) bb[0][j] = *(const v16bf*)(wbase + (size_t)j * 512);
    }
    for (int s = 0; s < KT; s += 2) {
        // stage 1 <- step s+1 (always valid: KT even)
        {
            const __bf16* p = aptr(s + 1);
            alo[1] = *(const v8bf*)p;
            ahi[1] = *(const v8bf*)(p + 16);
            const __bf16* bp = wbase + (size_t)(s + 1) * (COUT / 16) * 512;
#pragma unroll
            for (int j = 0; j < 4; ++j) bb[1][j] = *(const v16bf*)(bp + (size_t)j * 512);
        }
        // compute step s from stage 0
        {
            v16bf a;
#pragma unroll
            for (int t = 0; t < 8; ++t) { a[t] = alo[0][t]; a[t + 8] = ahi[0][t]; }
#pragma unroll
            for (int j = 0; j < 4; ++j) acc[j] = wmma_bf16(a, bb[0][j], acc[j]);
        }
        // stage 0 <- step s+2 (benign reload of s on the last pass)
        {
            int s2 = (s + 2 < KT) ? s + 2 : s;
            const __bf16* p = aptr(s2);
            alo[0] = *(const v8bf*)p;
            ahi[0] = *(const v8bf*)(p + 16);
            const __bf16* bp = wbase + (size_t)s2 * (COUT / 16) * 512;
#pragma unroll
            for (int j = 0; j < 4; ++j) bb[0][j] = *(const v16bf*)(bp + (size_t)j * 512);
        }
        // compute step s+1 from stage 1
        {
            v16bf a;
#pragma unroll
            for (int t = 0; t < 8; ++t) { a[t] = alo[1][t]; a[t + 8] = ahi[1][t]; }
#pragma unroll
            for (int j = 0; j < 4; ++j) acc[j] = wmma_bf16(a, bb[1][j], acc[j]);
        }
    }

    // C/D layout: lanes 0-15: M=r, N=lane; lanes 16-31: M=8+r, N=lane-16
    int nn_lo = ngrp * 64 + (lane & 15);
    int mrow = m0 + (lane >> 4) * 8;
#pragma unroll
    for (int j = 0; j < 4; ++j) {
        int nn = nn_lo + j * 16;
        float bn = bias[nn];
#pragma unroll
        for (int r = 0; r < 8; ++r) {
            float v = acc[j][r] + bn;
            int vx = mrow + r;
            if (MODE == 0) {
                outb[(size_t)vx * COUT + nn] = f2bf(v);
            } else {
                int xv = vx & 31, yv = (vx >> 5) & 31, zv = (vx >> 10) & 31, bb2 = vx >> 15;
                outf[(((size_t)bb2 * COUT + nn) << 15) + zv * 1024 + yv * 32 + xv] = v;
            }
        }
    }
}

// ---------------------------------------------------------------------------
// GroupNorm (groups=8), two-pass on bf16 ch-last tensors
// ---------------------------------------------------------------------------
__global__ __launch_bounds__(256) void gn_stats(const __bf16* __restrict__ x,
                                                float* __restrict__ stats,
                                                int Cc, int gsh) {
    int bg = blockIdx.x;            // b*8 + g  (32 values)
    int b = bg >> 3, g = bg & 7;
    int slab = blockIdx.y;          // 32 slabs of 1024 voxels
    int gch = 1 << gsh;
    int total = 1024 << gsh;
    float s = 0.f, q = 0.f;
    size_t voxbase = (size_t)b * R3 + slab * 1024;
    for (int i = threadIdx.x; i < total; i += 256) {
        int vox = i >> gsh;
        int ch = (g << gsh) + (i & (gch - 1));
        float v = bf2f(x[(voxbase + vox) * Cc + ch]);
        s += v; q += v * v;
    }
    __shared__ float ls[256], lq[256];
    ls[threadIdx.x] = s; lq[threadIdx.x] = q;
    __syncthreads();
    for (int st = 128; st > 0; st >>= 1) {
        if (threadIdx.x < st) {
            ls[threadIdx.x] += ls[threadIdx.x + st];
            lq[threadIdx.x] += lq[threadIdx.x + st];
        }
        __syncthreads();
    }
    if (threadIdx.x == 0) {
        atomicAdd(&stats[bg * 2 + 0], ls[0]);
        atomicAdd(&stats[bg * 2 + 1], lq[0]);
    }
}

__global__ __launch_bounds__(256) void gn_apply(__bf16* __restrict__ x,
                                                const float* __restrict__ stats,
                                                const float* __restrict__ gamma,
                                                const float* __restrict__ beta,
                                                int chShift) {
    size_t i = (size_t)blockIdx.x * 256 + threadIdx.x;
    int Cc = 1 << chShift;
    int ch = (int)(i & (Cc - 1));
    size_t rest = i >> chShift;            // b*32768 + vox
    int b = (int)(rest >> 15);
    int gsh = chShift - 3;
    int g = ch >> gsh;
    float cnt = 32768.f * (float)(1 << gsh);
    float s1 = stats[(b * 8 + g) * 2], s2 = stats[(b * 8 + g) * 2 + 1];
    float mu = s1 / cnt;
    float var = s2 / cnt - mu * mu;
    float v = bf2f(x[i]);
    v = (v - mu) * rsqrtf(var + 1e-5f) * gamma[ch] + beta[ch];
    x[i] = f2bf(gelu_f(v));
}

// ---------------------------------------------------------------------------
// z = mean + eps * exp(0.5*logvar); NCDHW fp32 inputs -> ch-last bf16
// ---------------------------------------------------------------------------
__global__ __launch_bounds__(256) void make_z(const float* __restrict__ mean,
                                              const float* __restrict__ lv,
                                              const float* __restrict__ eps,
                                              __bf16* __restrict__ zb) {
    size_t i = (size_t)blockIdx.x * 256 + threadIdx.x;    // over 4*128*32768
    int sp = (int)(i & 32767);
    int c  = (int)((i >> 15) & 127);
    int b  = (int)(i >> 22);
    float v = mean[i] + eps[i] * expf(0.5f * lv[i]);
    zb[(((size_t)b << 15) + sp) * Cch + c] = f2bf(v);
}

// ---------------------------------------------------------------------------
// Trilinear sample of zp (ch-last bf16) + positional embedding -> X0 (B*N,224)
// One wave per point: lane handles 4 channels, 3 PE values.
// ---------------------------------------------------------------------------
__global__ __launch_bounds__(256) void sample_pe(const float* __restrict__ pts,
                                                 const __bf16* __restrict__ zp,
                                                 __bf16* __restrict__ X0) {
    int wv = (int)((blockIdx.x * 256 + threadIdx.x) >> 5);   // point index
    int lane = threadIdx.x & 31;
    int b = wv >> 14;
    float px = pts[(size_t)wv * 3 + 0];
    float py = pts[(size_t)wv * 3 + 1];
    float pz = pts[(size_t)wv * 3 + 2];
    // pts_xyz = points[..., ::-1]: gx<-pz, gy<-py, gz<-px
    float gx = (pz + 1.f) * 0.5f * 31.f;
    float gy = (py + 1.f) * 0.5f * 31.f;
    float gz = (px + 1.f) * 0.5f * 31.f;
    float fx = floorf(gx), fy = floorf(gy), fz = floorf(gz);
    float wx = gx - fx, wy = gy - fy, wz = gz - fz;
    int xi0 = (int)fx, yi0 = (int)fy, zi0 = (int)fz;

    int c = lane * 4;
    float acc[4] = {0.f, 0.f, 0.f, 0.f};
#pragma unroll
    for (int cz = 0; cz < 2; ++cz)
#pragma unroll
        for (int cy = 0; cy < 2; ++cy)
#pragma unroll
            for (int cx = 0; cx < 2; ++cx) {
                int zi = zi0 + cz, yi = yi0 + cy, xi = xi0 + cx;
                float wgt = (cz ? wz : 1.f - wz) * (cy ? wy : 1.f - wy) * (cx ? wx : 1.f - wx);
                if (((unsigned)zi < 32u) && ((unsigned)yi < 32u) && ((unsigned)xi < 32u)) {
                    const __bf16* p =
                        zp + ((((size_t)b * 32 + zi) * 32 + yi) * 32 + xi) * Cch + c;
#pragma unroll
                    for (int j = 0; j < 4; ++j) acc[j] += wgt * bf2f(p[j]);
                }
            }
    __bf16* row = X0 + (size_t)wv * 224;
#pragma unroll
    for (int j = 0; j < 4; ++j) row[c + j] = f2bf(acc[j]);

    // pos embed: per coord 16 sin then 16 cos; 96 total, 3 per lane
    float coords[3] = {px, py, pz};
#pragma unroll
    for (int t = 0; t < 3; ++t) {
        int e = lane * 3 + t;              // 0..95
        int cc = e >> 5;
        int w = e & 31;
        float fr = expf((float)(w & 15) * (-9.2103403719761836f / 15.f));
        float ee = coords[cc] * fr;
        row[128 + e] = f2bf((w < 16) ? sinf(ee) : cosf(ee));
    }
}

// ---------------------------------------------------------------------------
// MLP GEMM: Y(M x N) = act(X(M x K) @ Wpk + bias), bf16 in/out, f32 accum.
// Wave: 16x64; block: 32 (M) x 256 (N). Fully unrolled K-loop (compiler
// renames registers per iteration -> pipelined loads, no rotation copies).
// ---------------------------------------------------------------------------
template <int K, int N, bool ACT>
__global__ __launch_bounds__(256) void gemm_wmma(const __bf16* __restrict__ X,
                                                 const __bf16* __restrict__ W,
                                                 const float* __restrict__ bias,
                                                 __bf16* __restrict__ Y) {
    constexpr int WN = N / 64;
    constexpr int KT = K / 32;
    int wave = threadIdx.x >> 5, lane = threadIdx.x & 31;
    int mtile = blockIdx.x * (8 / WN) + wave / WN;
    int ngrp  = wave % WN;
    int m0 = mtile * 16;
    int off = (lane >> 4) * 8;
    const __bf16* arow  = X + (size_t)(m0 + (lane & 15)) * K + off;
    const __bf16* wbase = W + (size_t)(ngrp * 4) * 512 + lane * 16;

    v8f acc[4] = {};
    v8bf alo = *(const v8bf*)arow;
    v8bf ahi = *(const v8bf*)(arow + 16);
    v16bf bc[4];
#pragma unroll
    for (int j = 0; j < 4; ++j) bc[j] = *(const v16bf*)(wbase + (size_t)j * 512);

#pragma unroll
    for (int s = 0; s < KT; ++s) {
        int sn = (s + 1 < KT) ? s + 1 : s;
        const __bf16* ap = arow + sn * 32;
        __builtin_prefetch(ap, 0, 3);
        v8bf nlo = *(const v8bf*)ap;
        v8bf nhi = *(const v8bf*)(ap + 16);
        const __bf16* bp = wbase + (size_t)sn * (N / 16) * 512;
        v16bf bn[4];
#pragma unroll
        for (int j = 0; j < 4; ++j) bn[j] = *(const v16bf*)(bp + (size_t)j * 512);

        v16bf a;
#pragma unroll
        for (int t = 0; t < 8; ++t) { a[t] = alo[t]; a[t + 8] = ahi[t]; }
#pragma unroll
        for (int j = 0; j < 4; ++j) acc[j] = wmma_bf16(a, bc[j], acc[j]);

        alo = nlo; ahi = nhi;
#pragma unroll
        for (int j = 0; j < 4; ++j) bc[j] = bn[j];
    }

    int nn_lo = ngrp * 64 + (lane & 15);
    int mrow = m0 + (lane >> 4) * 8;
#pragma unroll
    for (int j = 0; j < 4; ++j) {
        int nn = nn_lo + j * 16;
        float bn = bias[nn];
#pragma unroll
        for (int r = 0; r < 8; ++r) {
            float v = acc[j][r] + bn;
            if (ACT) v = gelu_f(v);
            Y[(size_t)(mrow + r) * N + nn] = f2bf(v);
        }
    }
}

// Final layer: (65536 x 256) @ (256 x 1) + b -> d_out[0:65536]
__global__ __launch_bounds__(256) void mlp_last(const __bf16* __restrict__ X,
                                                const float* __restrict__ w,
                                                const float* __restrict__ b,
                                                float* __restrict__ out) {
    int p = blockIdx.x * 256 + threadIdx.x;
    const __bf16* row = X + (size_t)p * 256;
    float s = b[0];
#pragma unroll 8
    for (int k = 0; k < 256; ++k) s += bf2f(row[k]) * w[k];
    out[p] = s;
}

// ---------------------------------------------------------------------------
// Host launcher
// ---------------------------------------------------------------------------
extern "C" void kernel_launch(void* const* d_in, const int* in_sizes, int n_in,
                              void* d_out, int out_size, void* d_ws, size_t ws_size,
                              hipStream_t stream) {
    (void)in_sizes; (void)n_in; (void)out_size; (void)ws_size;

    const float* points = (const float*)d_in[0];
    const float* sdfin  = (const float*)d_in[1];
    const float* eps    = (const float*)d_in[2];
    const float* pm_w0  = (const float*)d_in[3];
    const float* pm_b0  = (const float*)d_in[4];
    const float* pm_w1  = (const float*)d_in[5];
    const float* pm_b1  = (const float*)d_in[6];
    const float* ec1_w  = (const float*)d_in[7];
    const float* ec1_b  = (const float*)d_in[8];
    const float* eg1_g  = (const float*)d_in[9];
    const float* eg1_b  = (const float*)d_in[10];
    const float* ec2_w  = (const float*)d_in[11];
    const float* ec2_b  = (const float*)d_in[12];
    const float* eg2_g  = (const float*)d_in[13];
    const float* eg2_b  = (const float*)d_in[14];
    const float* mean_w = (const float*)d_in[15];
    const float* mean_b = (const float*)d_in[16];
    const float* lv_w   = (const float*)d_in[17];
    const float* lv_b   = (const float*)d_in[18];
    const float* dc1_w  = (const float*)d_in[19];
    const float* dc1_b  = (const float*)d_in[20];
    const float* dg1_g  = (const float*)d_in[21];
    const float* dg1_b  = (const float*)d_in[22];
    const float* dc2_w  = (const float*)d_in[23];
    const float* dc2_b  = (const float*)d_in[24];
    const float* dg2_g  = (const float*)d_in[25];
    const float* dg2_b  = (const float*)d_in[26];
    const float* mlp_w[6]; const float* mlp_b[6];
    for (int i = 0; i < 6; ++i) {
        mlp_w[i] = (const float*)d_in[27 + 2 * i];
        mlp_b[i] = (const float*)d_in[28 + 2 * i];
    }

    float* out      = (float*)d_out;
    float* out_sdf  = out;                       // 65536
    float* out_mean = out + NPTOT;               // 4*128*32768
    float* out_lv   = out_mean + (size_t)Bsz * Cch * R3;

    // ---- workspace carve-up (with reuse) ----
    char* ws = (char*)d_ws;
    size_t off = 0;
    auto alloc = [&](size_t bytes) -> void* {
        void* p = ws + off;
        off += (bytes + 255) & ~(size_t)255;
        return p;
    };
    float*  grid   = (float*)alloc((size_t)Bsz * R3 * Cch * 4);      // 64 MB
    __bf16* bufA   = (__bf16*)alloc((size_t)Bsz * R3 * Cch * 2);     // act0 / zb
    __bf16* bufB   = (__bf16*)alloc((size_t)Bsz * R3 * 2 * Cch * 2); // e1 / zp
    __bf16* bufC   = (__bf16*)alloc((size_t)Bsz * R3 * 2 * Cch * 2); // e2 / X0
    __bf16* bufD   = (__bf16*)alloc((size_t)NPTOT * 256 * 2);        // d1 / H1
    __bf16* bufE   = (__bf16*)alloc((size_t)NPTOT * 256 * 2);        // H2
    __bf16* pk_ec1 = (__bf16*)alloc((size_t)27 * 128 * 256 * 2);
    __bf16* pk_ec2 = (__bf16*)alloc((size_t)27 * 256 * 256 * 2);
    __bf16* pk_mw  = (__bf16*)alloc((size_t)27 * 256 * 128 * 2);
    __bf16* pk_lw  = (__bf16*)alloc((size_t)27 * 256 * 128 * 2);
    __bf16* pk_dc1 = (__bf16*)alloc((size_t)27 * 128 * 128 * 2);
    __bf16* pk_dc2 = (__bf16*)alloc((size_t)27 * 128 * 128 * 2);
    __bf16* pk_m0  = (__bf16*)alloc((size_t)224 * 256 * 2);
    __bf16* pk_m1  = (__bf16*)alloc((size_t)256 * 256 * 2);
    __bf16* pk_m2  = (__bf16*)alloc((size_t)256 * 256 * 2);
    __bf16* pk_m3  = (__bf16*)alloc((size_t)256 * 256 * 2);
    __bf16* pk_m4  = (__bf16*)alloc((size_t)256 * 256 * 2);
    float*  stats  = (float*)alloc(4 * 32 * 2 * sizeof(float));
    __bf16* zpage  = (__bf16*)alloc(256);        // branch-free halo zero page

    __bf16* act0 = bufA;  __bf16* zb = bufA;
    __bf16* e1   = bufB;  __bf16* zp = bufB;
    __bf16* e2   = bufC;  __bf16* X0 = bufC;
    __bf16* d1   = bufD;  __bf16* H1 = bufD;
    __bf16* H2   = bufE;

    // ---- init ----
    hipMemsetAsync(grid, 0, (size_t)Bsz * R3 * Cch * 4, stream);
    hipMemsetAsync(stats, 0, 4 * 32 * 2 * sizeof(float), stream);
    hipMemsetAsync(zpage, 0, 256, stream);

    // ---- weight repacks ----
    auto cgrid = [](size_t n) { return (unsigned)((n + 255) / 256); };
    repack_conv_w<<<cgrid((size_t)27 * 128 * 256), 256, 0, stream>>>(ec1_w, pk_ec1, 128, 256);
    repack_conv_w<<<cgrid((size_t)27 * 256 * 256), 256, 0, stream>>>(ec2_w, pk_ec2, 256, 256);
    repack_conv_w<<<cgrid((size_t)27 * 256 * 128), 256, 0, stream>>>(mean_w, pk_mw, 256, 128);
    repack_conv_w<<<cgrid((size_t)27 * 256 * 128), 256, 0, stream>>>(lv_w, pk_lw, 256, 128);
    repack_conv_w<<<cgrid((size_t)27 * 128 * 128), 256, 0, stream>>>(dc1_w, pk_dc1, 128, 128);
    repack_conv_w<<<cgrid((size_t)27 * 128 * 128), 256, 0, stream>>>(dc2_w, pk_dc2, 128, 128);
    repack_mlp_w<<<cgrid((size_t)224 * 256), 256, 0, stream>>>(mlp_w[0], pk_m0, 224, 256);
    repack_mlp_w<<<cgrid((size_t)256 * 256), 256, 0, stream>>>(mlp_w[1], pk_m1, 256, 256);
    repack_mlp_w<<<cgrid((size_t)256 * 256), 256, 0, stream>>>(mlp_w[2], pk_m2, 256, 256);
    repack_mlp_w<<<cgrid((size_t)256 * 256), 256, 0, stream>>>(mlp_w[3], pk_m3, 256, 256);
    repack_mlp_w<<<cgrid((size_t)256 * 256), 256, 0, stream>>>(mlp_w[4], pk_m4, 256, 256);

    // ---- point MLP + scatter, convert grid to bf16 ----
    point_mlp_scatter<<<NPTOT / 256, 256, 0, stream>>>(points, sdfin, pm_w0, pm_b0,
                                                       pm_w1, pm_b1, grid);
    cvt_f32_bf16<<<cgrid((size_t)Bsz * R3 * Cch), 256, 0, stream>>>(
        grid, act0, (size_t)Bsz * R3 * Cch);

    // ---- encoder ----
    conv3d_wmma<128, 256, 0><<<4096, 256, 0, stream>>>(act0, pk_ec1, ec1_b, zpage, e1, nullptr);
    gn_stats<<<dim3(32, 32), 256, 0, stream>>>(e1, stats + 0, 256, 5);
    gn_apply<<<131072, 256, 0, stream>>>(e1, stats + 0, eg1_g, eg1_b, 8);
    conv3d_wmma<256, 256, 0><<<4096, 256, 0, stream>>>(e1, pk_ec2, ec2_b, zpage, e2, nullptr);
    gn_stats<<<dim3(32, 32), 256, 0, stream>>>(e2, stats + 64, 256, 5);
    gn_apply<<<131072, 256, 0, stream>>>(e2, stats + 64, eg2_g, eg2_b, 8);
    conv3d_wmma<256, 128, 1><<<2048, 256, 0, stream>>>(e2, pk_mw, mean_b, zpage, nullptr, out_mean);
    conv3d_wmma<256, 128, 1><<<2048, 256, 0, stream>>>(e2, pk_lw, lv_b, zpage, nullptr, out_lv);

    // ---- reparameterize + decoder ----
    make_z<<<65536, 256, 0, stream>>>(out_mean, out_lv, eps, zb);
    conv3d_wmma<128, 128, 0><<<2048, 256, 0, stream>>>(zb, pk_dc1, dc1_b, zpage, d1, nullptr);
    gn_stats<<<dim3(32, 32), 256, 0, stream>>>(d1, stats + 128, 128, 4);
    gn_apply<<<65536, 256, 0, stream>>>(d1, stats + 128, dg1_g, dg1_b, 7);
    conv3d_wmma<128, 128, 0><<<2048, 256, 0, stream>>>(d1, pk_dc2, dc2_b, zpage, zp, nullptr);
    gn_stats<<<dim3(32, 32), 256, 0, stream>>>(zp, stats + 192, 128, 4);
    gn_apply<<<65536, 256, 0, stream>>>(zp, stats + 192, dg2_g, dg2_b, 7);

    // ---- sample + pos-embed + point MLP head ----
    sample_pe<<<8192, 256, 0, stream>>>(points, zp, X0);
    gemm_wmma<224, 256, true><<<2048, 256, 0, stream>>>(X0, pk_m0, mlp_b[0], H1);
    gemm_wmma<256, 256, true><<<2048, 256, 0, stream>>>(H1, pk_m1, mlp_b[1], H2);
    gemm_wmma<256, 256, true><<<2048, 256, 0, stream>>>(H2, pk_m2, mlp_b[2], H1);
    gemm_wmma<256, 256, true><<<2048, 256, 0, stream>>>(H1, pk_m3, mlp_b[3], H2);
    gemm_wmma<256, 256, true><<<2048, 256, 0, stream>>>(H2, pk_m4, mlp_b[4], H1);
    mlp_last<<<NPTOT / 256, 256, 0, stream>>>(H1, mlp_w[5], mlp_b[5], out_sdf);
}
